// MSSPM_81810537054218
// MI455X (gfx1250) — compile-verified, matrix operand order checked by reading, not run on hardware
//
#include <hip/hip_runtime.h>
#include <hip/hip_bf16.h>
#include <math.h>

// Problem dims (fixed by reference)
#define BB  32
#define LL  512
#define DD  1024
#define HH  256
#define EE  64
#define G4H 1024   // 4*H
#define H2  512    // 2*H

typedef __attribute__((ext_vector_type(16))) __bf16 v16bf;
typedef __attribute__((ext_vector_type(8)))  float  v8f;

union ABf { v16bf v; unsigned int u[8]; };
union CDf { v8f   v; float        f[8]; };

__device__ __forceinline__ unsigned short f2bf(float f) {
  unsigned int x = __float_as_uint(f);
  x += 0x7FFFu + ((x >> 16) & 1u);          // round-to-nearest-even
  return (unsigned short)(x >> 16);
}
__device__ __forceinline__ float sigmf(float x) { return 1.0f / (1.0f + __expf(-x)); }

// K-offset within a 32-wide K chunk for 16-bit A/B WMMA fragments (wave32):
// lanes 0-15 hold K {0..7,16..23}, lanes 16-31 hold K {8..15,24..31}, 2 per dword.
__device__ __forceinline__ int klay(int lane, int v) {
  return ((lane & 16) ? 8 : 0) + ((v & 4) ? 16 : 0) + (v & 3) * 2;
}

// ---------------------------------------------------------------- converts --
__global__ void k_cvt_bf16(const float* __restrict__ src,
                           unsigned short* __restrict__ dst, size_t n) {
  size_t i = (size_t)blockIdx.x * blockDim.x + threadIdx.x;
  if (i < n) dst[i] = f2bf(src[i]);
}

// x (B,L,D) f32 -> time-major bf16 (L*B, D)
__global__ void k_x_tmajor(const float* __restrict__ x,
                           unsigned short* __restrict__ dst) {
  size_t i = (size_t)blockIdx.x * blockDim.x + threadIdx.x;
  const size_t total = (size_t)LL * BB * DD;
  if (i >= total) return;
  int d = (int)(i % DD);
  size_t r = i / DD;
  int b = (int)(r % BB);
  int t = (int)(r / BB);
  dst[i] = f2bf(x[((size_t)b * LL + t) * DD + d]);
}

// Permute gate rows [i;f;g;o] (4H,K) -> interleaved rows n=4j+q, convert bf16.
__global__ void k_perm_w(const float* __restrict__ w,
                         unsigned short* __restrict__ dst, int K) {
  size_t i = (size_t)blockIdx.x * blockDim.x + threadIdx.x;
  size_t total = (size_t)G4H * K;
  if (i >= total) return;
  int k = (int)(i % K);
  int n = (int)(i / K);
  int j = n >> 2, q = n & 3;
  dst[(size_t)n * K + k] = f2bf(w[((size_t)(q * HH + j)) * K + k]);
}

__global__ void k_perm_b(const float* __restrict__ b, float* __restrict__ dst) {
  int n = blockIdx.x * blockDim.x + threadIdx.x;
  if (n >= G4H) return;
  dst[n] = b[(n & 3) * HH + (n >> 2)];
}

// ----------------------------------------------------------------- WMMA GEMM
// C[M,N] = A[M,K](bf16) * W[N,K](bf16)^T + bias[N]; block=256 (8 waves),
// block tile 128x64, wave tile 32x32 (2x2 WMMA 16x16x32 bf16 tiles).
__global__ __launch_bounds__(256) void
k_gemm(const unsigned short* __restrict__ A, const unsigned short* __restrict__ W,
       const float* __restrict__ bias, float* __restrict__ C,
       int M, int N, int K) {
  const int lane = threadIdx.x & 31;
  const int w    = threadIdx.x >> 5;
  const int m0 = blockIdx.x * 128 + (w & 3) * 32;
  const int n0 = blockIdx.y * 64  + (w >> 2) * 32;
  const int cL = lane & 15;
  const int hi8 = (lane & 16) ? 8 : 0;

  CDf acc[2][2];
#pragma unroll
  for (int mi = 0; mi < 2; ++mi)
#pragma unroll
    for (int ni = 0; ni < 2; ++ni)
#pragma unroll
      for (int r = 0; r < 8; ++r) acc[mi][ni].f[r] = 0.0f;

  for (int kb = 0; kb < K; kb += 32) {
    ABf a[2], bb[2];
#pragma unroll
    for (int mi = 0; mi < 2; ++mi) {
      size_t m = (size_t)(m0 + mi * 16 + cL);
#pragma unroll
      for (int v = 0; v < 8; ++v)
        a[mi].u[v] = *(const unsigned int*)(A + m * K + kb + klay(lane, v));
    }
#pragma unroll
    for (int ni = 0; ni < 2; ++ni) {
      size_t n = (size_t)(n0 + ni * 16 + cL);
#pragma unroll
      for (int v = 0; v < 8; ++v)
        bb[ni].u[v] = *(const unsigned int*)(W + n * K + kb + klay(lane, v));
    }
    if (kb + 32 < K) {  // pull next K-chunk lines toward WGP$ (global_prefetch_b8)
      __builtin_prefetch(A + (size_t)(m0 + cL) * K + kb + 32, 0, 1);
      __builtin_prefetch(W + (size_t)(n0 + cL) * K + kb + 32, 0, 1);
    }
#pragma unroll
    for (int mi = 0; mi < 2; ++mi)
#pragma unroll
      for (int ni = 0; ni < 2; ++ni)
        acc[mi][ni].v = __builtin_amdgcn_wmma_f32_16x16x32_bf16(
            false, a[mi].v, false, bb[ni].v, (short)0, acc[mi][ni].v, false, false);
  }

#pragma unroll
  for (int mi = 0; mi < 2; ++mi)
#pragma unroll
    for (int ni = 0; ni < 2; ++ni) {
      int col = n0 + ni * 16 + cL;
      float bv = bias ? bias[col] : 0.0f;
#pragma unroll
      for (int r = 0; r < 8; ++r) {
        int row = m0 + mi * 16 + r + hi8;
        C[(size_t)row * N + col] = acc[mi][ni].f[r] + bv;
      }
    }
}

// -------------------------------------------------------- LSTM recurrence ---
// grid = 2 (direction), block = 1024 (32 waves). Per step: D = xg + h*Whh^T
// (32x1024, K=256) entirely via WMMA; gate-interleaved columns (4j+q) so the
// nonlinearity is intra-wave (__shfl over 4-lane groups). h lives in LDS
// (bf16, A-matrix source), c lives in VGPRs. The per-step xg seed tile
// (128 KB, cold in HBM) is double-buffered in LDS via CDNA5 async
// global->LDS copies (ASYNCcnt), issued one step ahead so HBM latency is
// hidden behind the WMMA chain of the current step.
#define HS   264   // padded h row stride (bf16 units)
#define XGS  1032  // padded xg row stride (f32), 1032 % 64 = 8 -> no bank dup
__global__ __launch_bounds__(1024) void
k_lstm(const float* __restrict__ xg_f, const float* __restrict__ xg_r,
       const unsigned short* __restrict__ whh_f,
       const unsigned short* __restrict__ whh_r,
       const int* __restrict__ lens, float* __restrict__ out) {
  const int dir = blockIdx.x;
  const float* __restrict__ xg = dir ? xg_r : xg_f;
  const unsigned short* __restrict__ whh = dir ? whh_r : whh_f;

  __shared__ float          s_xg[2][32 * XGS];   // 2 x 129 KB staging
  __shared__ unsigned short s_h[32 * HS];
  __shared__ int            s_lens[BB];

  const int lane = threadIdx.x & 31;
  const int w    = threadIdx.x >> 5;
  const int m0   = (w >> 4) * 16;     // M tile: rows 0..15 or 16..31
  const int ntb  = (w & 15) * 4;      // 4 consecutive N tiles per wave
  const int cL   = lane & 15;
  const int hi8  = (lane & 16) ? 8 : 0;
  const int base = lane & ~3;         // 4-lane gate group leader

  if (threadIdx.x < BB) s_lens[threadIdx.x] = lens[threadIdx.x];
  for (int i = threadIdx.x; i < 32 * HS; i += 1024) s_h[i] = 0;

  float c[4][8];
#pragma unroll
  for (int nt = 0; nt < 4; ++nt)
#pragma unroll
    for (int r = 0; r < 8; ++r) c[nt][r] = 0.0f;

  __syncthreads();  // s_lens + zeroed s_h visible before any async gather

  // Issue the (gathered) xg tile for `step` into LDS buffer `buf`:
  // 8192 x 16B units, 8 per thread. Reverse dir gathers t = len-1-step.
  auto issue_xg = [&](int step, int buf) {
    unsigned lbase = (unsigned)(size_t)(&s_xg[buf][0]);
#pragma unroll
    for (int cp = 0; cp < 8; ++cp) {
      int u  = (int)threadIdx.x + cp * 1024;  // 16B unit id
      int b  = u >> 8;                        // 256 units per row
      int cu = u & 255;                       // 16B column unit
      int len = s_lens[b];
      int tin = dir ? ((step < len) ? (len - 1 - step) : step) : step;
      const float* gp = xg + ((size_t)tin * BB + b) * G4H + cu * 4;
      unsigned lo = lbase + (unsigned)((b * XGS + cu * 4) * 4);
      asm volatile("global_load_async_to_lds_b128 %0, %1, off"
                   :: "v"(lo), "v"((unsigned long long)(size_t)gp)
                   : "memory");
    }
  };

  issue_xg(0, 0);
  asm volatile("s_wait_asynccnt 0x0" ::: "memory");
  __syncthreads();

  for (int s = 0; s < LL; ++s) {
    const int cur = s & 1;
    if (s + 1 < LL) issue_xg(s + 1, cur ^ 1);   // prefetch next step's seed

    CDf acc[4];
#pragma unroll
    for (int nt = 0; nt < 4; ++nt) {            // seed from LDS (bias folded)
      int n = (ntb + nt) * 16 + cL;
#pragma unroll
      for (int r = 0; r < 8; ++r) {
        int b = m0 + r + hi8;
        acc[nt].f[r] = s_xg[cur][b * XGS + n];
      }
    }
    // h * Whh^T : 8 K-chunks x 4 N-tiles of v_wmma_f32_16x16x32_bf16
#pragma unroll
    for (int kb = 0; kb < HH; kb += 32) {
      ABf a;
      int mrow = m0 + cL;
#pragma unroll
      for (int v = 0; v < 8; ++v)
        a.u[v] = *(const unsigned int*)&s_h[mrow * HS + kb + klay(lane, v)];
#pragma unroll
      for (int nt = 0; nt < 4; ++nt) {
        ABf bf;
        size_t n = (size_t)((ntb + nt) * 16 + cL);
#pragma unroll
        for (int v = 0; v < 8; ++v)
          bf.u[v] = *(const unsigned int*)(whh + n * HH + kb + klay(lane, v));
        acc[nt].v = __builtin_amdgcn_wmma_f32_16x16x32_bf16(
            false, a.v, false, bf.v, (short)0, acc[nt].v, false, false);
      }
    }
    __syncthreads();  // all waves done reading s_h for this step

    // gates: columns 4j+{0..3} = i,f,g,o live in one 4-lane group
#pragma unroll
    for (int nt = 0; nt < 4; ++nt) {
      int junit = (ntb + nt) * 4 + (cL >> 2);
#pragma unroll
      for (int r = 0; r < 8; ++r) {
        float vi = __shfl(acc[nt].f[r], base + 0, 32);
        float vf = __shfl(acc[nt].f[r], base + 1, 32);
        float vg = __shfl(acc[nt].f[r], base + 2, 32);
        float vo = __shfl(acc[nt].f[r], base + 3, 32);
        float cc = sigmf(vf) * c[nt][r] + sigmf(vi) * tanhf(vg);
        c[nt][r] = cc;
        float h = sigmf(vo) * tanhf(cc);
        if ((lane & 3) == 0) {
          int m = m0 + r + hi8;
          s_h[m * HS + junit] = f2bf(h);
          int len = s_lens[m];
          int tout; float val;
          if (dir) {
            if (s < len) { tout = len - 1 - s; val = h; }
            else         { tout = s;           val = 0.0f; }
          } else {
            tout = s; val = (s < len) ? h : 0.0f;   // pad_packed zeroing
          }
          out[((size_t)tout * BB + m) * H2 + dir * HH + junit] = val;
        }
      }
    }
    asm volatile("s_wait_asynccnt 0x0" ::: "memory");  // next seed resident
    __syncthreads();  // s_h + s_xg[cur^1] ready for next step
  }
}

// ------------------------------------------------------------------- CRF ----
__global__ __launch_bounds__(64) void k_softmax(float* __restrict__ e) {
  __shared__ float red[64];
  int row = blockIdx.x, tid = threadIdx.x;
  float v = e[(size_t)row * EE + tid];
  red[tid] = v; __syncthreads();
  for (int o = 32; o > 0; o >>= 1) { if (tid < o) red[tid] = fmaxf(red[tid], red[tid + o]); __syncthreads(); }
  float mx = red[0]; __syncthreads();
  float ex = __expf(v - mx);
  red[tid] = ex; __syncthreads();
  for (int o = 32; o > 0; o >>= 1) { if (tid < o) red[tid] += red[tid + o]; __syncthreads(); }
  e[(size_t)row * EE + tid] = ex / red[0];
}

__global__ void k_score(const float* __restrict__ emit, const int* __restrict__ tags,
                        const int* __restrict__ lens, const float* __restrict__ trans,
                        float* __restrict__ totals) {
  int b = threadIdx.x;
  if (b >= BB) return;
  int len = lens[b];
  float total = 0.0f; int prev = 0;
  for (int t = 0; t < len; ++t) {
    int tg = tags[(size_t)b * LL + t];
    float sc = emit[((size_t)t * BB + b) * EE + tg];
    if (t > 0) sc += trans[prev * EE + tg];
    total += sc; prev = tg;
  }
  totals[b] = total;
}

__global__ __launch_bounds__(64) void
k_crf(const float* __restrict__ emit, const float* __restrict__ trans,
      const int* __restrict__ lens, const float* __restrict__ totals,
      float* __restrict__ loss_b) {
  __shared__ float st[EE * EE];
  __shared__ float d[EE];
  __shared__ float red[EE];
  int b = blockIdx.x, j = threadIdx.x;
  for (int i = j; i < EE * EE; i += 64) st[i] = trans[i];
  d[j] = emit[((size_t)0 * BB + b) * EE + j];
  __syncthreads();
  int len = lens[b];
  for (int t = 1; t < len; ++t) {
    float m = -1e30f, s = 0.0f;
#pragma unroll 4
    for (int k = 0; k < EE; ++k) {           // online logsumexp over prev state
      float v = d[k] + st[k * EE + j];
      if (v > m) { s = s * __expf(m - v) + 1.0f; m = v; }
      else       { s += __expf(v - m); }
    }
    float dn = m + __logf(s) + emit[((size_t)t * BB + b) * EE + j];
    __syncthreads();
    d[j] = dn;
    __syncthreads();
  }
  red[j] = d[j]; __syncthreads();
  for (int o = 32; o > 0; o >>= 1) { if (j < o) red[j] = fmaxf(red[j], red[j + o]); __syncthreads(); }
  float mx = red[0]; __syncthreads();
  red[j] = __expf(d[j] - mx); __syncthreads();
  for (int o = 32; o > 0; o >>= 1) { if (j < o) red[j] += red[j + o]; __syncthreads(); }
  if (j == 0) {
    float Z = mx + __logf(red[0]);
    loss_b[b] = -(totals[b] - Z) / (float)len;
  }
}

__global__ void k_final(const float* __restrict__ loss_b, float* __restrict__ out) {
  float s = 0.0f;
  for (int b = 0; b < BB; ++b) s += loss_b[b];
  out[0] = s / (float)BB;
}

// ------------------------------------------------------------------- host ---
#define CEILDIV(a, b) (((a) + (b) - 1) / (b))

extern "C" void kernel_launch(void* const* d_in, const int* in_sizes, int n_in,
                              void* d_out, int out_size, void* d_ws, size_t ws_size,
                              hipStream_t stream) {
  (void)in_sizes; (void)n_in; (void)out_size; (void)ws_size;
  const float* x        = (const float*)d_in[0];
  const int*   lens     = (const int*)d_in[1];
  const int*   tags     = (const int*)d_in[2];
  /* d_in[3] = mask (bool) — redundant with lens, unused */
  const float* w_ih_l0  = (const float*)d_in[4];
  const float* w_hh_l0  = (const float*)d_in[5];
  const float* b_l0     = (const float*)d_in[6];
  const float* w_ih_l12 = (const float*)d_in[7];
  const float* w_hh_l12 = (const float*)d_in[8];
  const float* b_l12    = (const float*)d_in[9];
  const float* fc_w     = (const float*)d_in[10];
  const float* fc_b     = (const float*)d_in[11];
  const float* trans    = (const float*)d_in[12];

  char* ws = (char*)d_ws;
  size_t off = 0;
  auto alloc = [&](size_t bytes) -> void* {
    void* p = ws + off;
    off = (off + bytes + 255) & ~(size_t)255;
    return p;
  };

  unsigned short* wih_bf[3][2];
  unsigned short* whh_bf[3][2];
  float*          bias_p[3][2];
  for (int li = 0; li < 3; ++li) {
    size_t K = (li == 0) ? DD : H2;
    for (int d = 0; d < 2; ++d) {
      wih_bf[li][d] = (unsigned short*)alloc((size_t)G4H * K * 2);
      whh_bf[li][d] = (unsigned short*)alloc((size_t)G4H * HH * 2);
      bias_p[li][d] = (float*)alloc(G4H * 4);
    }
  }
  unsigned short* fcw_bf = (unsigned short*)alloc((size_t)EE * H2 * 2);
  unsigned short* xa     = (unsigned short*)alloc((size_t)LL * BB * DD * 2);
  float* xg_f   = (float*)alloc((size_t)LL * BB * G4H * 4);
  float* xg_r   = (float*)alloc((size_t)LL * BB * G4H * 4);
  float* hs     = (float*)alloc((size_t)LL * BB * H2 * 4);
  float* emit   = (float*)alloc((size_t)LL * BB * EE * 4);
  float* totals = (float*)alloc(BB * 4);
  float* loss_b = (float*)alloc(BB * 4);

  // --- weight prep (bf16 + gate interleave) ---
  for (int d = 0; d < 2; ++d) {
    k_perm_w<<<CEILDIV((size_t)G4H * DD, 256), 256, 0, stream>>>(
        w_ih_l0 + (size_t)d * G4H * DD, wih_bf[0][d], DD);
    k_perm_w<<<CEILDIV((size_t)G4H * HH, 256), 256, 0, stream>>>(
        w_hh_l0 + (size_t)d * G4H * HH, whh_bf[0][d], HH);
    k_perm_b<<<CEILDIV(G4H, 256), 256, 0, stream>>>(b_l0 + (size_t)d * G4H, bias_p[0][d]);
    for (int li = 0; li < 2; ++li) {
      k_perm_w<<<CEILDIV((size_t)G4H * H2, 256), 256, 0, stream>>>(
          w_ih_l12 + ((size_t)li * 2 + d) * G4H * H2, wih_bf[li + 1][d], H2);
      k_perm_w<<<CEILDIV((size_t)G4H * HH, 256), 256, 0, stream>>>(
          w_hh_l12 + ((size_t)li * 2 + d) * G4H * HH, whh_bf[li + 1][d], HH);
      k_perm_b<<<CEILDIV(G4H, 256), 256, 0, stream>>>(
          b_l12 + ((size_t)li * 2 + d) * G4H, bias_p[li + 1][d]);
    }
  }
  k_cvt_bf16<<<CEILDIV((size_t)EE * H2, 256), 256, 0, stream>>>(fc_w, fcw_bf, (size_t)EE * H2);

  // --- BiLSTM stack ---
  k_x_tmajor<<<CEILDIV((size_t)LL * BB * DD, 256), 256, 0, stream>>>(x, xa);
  const int M = LL * BB;
  for (int li = 0; li < 3; ++li) {
    int K = (li == 0) ? DD : H2;
    if (li > 0)
      k_cvt_bf16<<<CEILDIV((size_t)M * H2, 256), 256, 0, stream>>>(hs, xa, (size_t)M * H2);
    dim3 gg(M / 128, G4H / 64);
    k_gemm<<<gg, 256, 0, stream>>>(xa, wih_bf[li][0], bias_p[li][0], xg_f, M, G4H, K);
    k_gemm<<<gg, 256, 0, stream>>>(xa, wih_bf[li][1], bias_p[li][1], xg_r, M, G4H, K);
    k_lstm<<<2, 1024, 0, stream>>>(xg_f, xg_r, whh_bf[li][0], whh_bf[li][1], lens, hs);
  }

  // --- CRF head ---
  k_cvt_bf16<<<CEILDIV((size_t)M * H2, 256), 256, 0, stream>>>(hs, xa, (size_t)M * H2);
  dim3 gfc(M / 128, 1);
  k_gemm<<<gfc, 256, 0, stream>>>(xa, fcw_bf, fc_b, emit, M, EE, H2);
  k_softmax<<<M, 64, 0, stream>>>(emit);
  k_score<<<1, 32, 0, stream>>>(emit, tags, lens, trans, totals);
  k_crf<<<BB, 64, 0, stream>>>(emit, trans, lens, totals, loss_b);
  k_final<<<1, 1, 0, stream>>>(loss_b, (float*)d_out);
}
